// EdgeCNN_33998961115945
// MI455X (gfx1250) — matrix operand order repaired and checked by reading, not compiled, not run on hardware
//
#include <hip/hip_runtime.h>
#include <hip/hip_bf16.h>

typedef _Float16 v16h __attribute__((ext_vector_type(16)));
typedef float    v8f  __attribute__((ext_vector_type(8)));

#define BN_INVC 0.99999500003749968f
#define NEG_SLOPE 0.2f

// ---------------- monotonic float<->uint encoding for atomic max ----------------
__device__ __forceinline__ unsigned enc_f(float f) {
    unsigned u = __float_as_uint(f);
    return (u & 0x80000000u) ? ~u : (u | 0x80000000u);
}
__device__ __forceinline__ float dec_f(unsigned e) {
    unsigned u = (e & 0x80000000u) ? (e & 0x7fffffffu) : ~e;
    return __uint_as_float(u);
}
__device__ __forceinline__ float lrelu(float v) { return v > 0.f ? v : NEG_SLOPE * v; }

// ---------------- wave32 top-20 over 1024 LDS floats (one wave per row) --------
__device__ __forceinline__ void wave_topk20(const float* distRow, int* outIdx) {
    const int lane = threadIdx.x & 31;
    float v[32];
#pragma unroll
    for (int i = 0; i < 32; ++i) v[i] = distRow[lane + 32 * i];
    for (int r = 0; r < 20; ++r) {
        float bm = -3.4e38f; int bi = 0;
#pragma unroll
        for (int i = 0; i < 32; ++i) { if (v[i] > bm) { bm = v[i]; bi = i; } }
        float m = bm;
        int cand = lane + (bi << 5);
#pragma unroll
        for (int off = 16; off > 0; off >>= 1) {
            float om = __shfl_xor(m, off, 32);
            int   oc = __shfl_xor(cand, off, 32);
            if (om > m || (om == m && oc < cand)) { m = om; cand = oc; }
        }
        if (lane == 0) outIdx[r] = cand;
        if ((cand & 31) == lane) {
            const int reg = cand >> 5;
#pragma unroll
            for (int i = 0; i < 32; ++i) if (i == reg) v[i] = -3.4e38f;
        }
    }
}

// ---------------- A-operand (16x32 f16) packed coordinates ---------------------
// element (m, klocal) -> lane = m + 16*(klocal>>4), slot = klocal & 15
// ---------------- B-operand (32x16 f16) packed coordinates ---------------------
// element (klocal, n) -> lane = n + 16*(klocal>>4), slot = klocal & 15

// ---------------- weight repack: W[o][k] (f32) -> B-packed f16 ----------------
__device__ void pack_one(const float* __restrict__ W, _Float16* __restrict__ dst,
                         int Cout, int K, int tid, int stride) {
    const int KB = K >> 5;
    for (int e = tid; e < Cout * K; e += stride) {
        int o = e / K, k = e % K;
        int nt = o >> 4, nl = o & 15, kb = k >> 5, kl = k & 31;
        int lane = nl + ((kl >> 4) << 4), slot = kl & 15;
        dst[(((size_t)(nt * KB + kb) * 32) + lane) * 16 + slot] = (_Float16)W[(size_t)o * K + k];
    }
}

__global__ void pack_weights_kernel(const float* W2, const float* W3, const float* W4, const float* W5,
                                    _Float16* p2, _Float16* p3, _Float16* p4, _Float16* p5) {
    int tid = blockIdx.x * blockDim.x + threadIdx.x;
    int stride = gridDim.x * blockDim.x;
    pack_one(W2, p2, 64, 128, tid, stride);
    pack_one(W3, p3, 128, 128, tid, stride);
    pack_one(W4, p4, 256, 256, tid, stride);
    pack_one(W5, p5, 512, 512, tid, stride);
}

__global__ void init_pooled_kernel(unsigned* pooled) {
    int t = blockIdx.x * blockDim.x + threadIdx.x;
    if (t < 8 * 512) pooled[t] = 0u;  // encodes below any real float
}

// ---------------- layer-1 KNN on 3-dim coords (scalar distances) ---------------
__global__ __launch_bounds__(256) void knn3_kernel(const float* __restrict__ x, int* __restrict__ idxout) {
    const int b = blockIdx.y, qbase = blockIdx.x * 16;
    const int tid = threadIdx.x, wave = tid >> 5;
    __shared__ float dist[16][1024];
    __shared__ float q[16][3];
    const float* xb = x + (size_t)b * 1024 * 3;
    if (tid < 48) ((float*)q)[tid] = xb[qbase * 3 + tid];
    __syncthreads();
    for (int e = tid; e < 16 * 1024; e += 256) {
        int m = e >> 10, c = e & 1023;
        float dx = q[m][0] - xb[c * 3 + 0];
        float dy = q[m][1] - xb[c * 3 + 1];
        float dz = q[m][2] - xb[c * 3 + 2];
        dist[m][c] = -(dx * dx + dy * dy + dz * dz);
    }
    __syncthreads();
    for (int rr = wave; rr < 16; rr += 8)
        wave_topk20(&dist[rr][0], idxout + ((size_t)b * 1024 + qbase + rr) * 20);
}

// ---------------- layer-1 EdgeConv (K=6, scalar) -------------------------------
__global__ __launch_bounds__(256) void edgeconv1_kernel(const float* __restrict__ x, const int* __restrict__ idx,
                                                        const float* __restrict__ W1, const float* __restrict__ g1,
                                                        const float* __restrict__ b1, float* __restrict__ x1) {
    const int b = blockIdx.y;
    const int p = blockIdx.x * 4 + (threadIdx.x >> 6);
    const int o = threadIdx.x & 63;
    const float* xb = x + (size_t)b * 1024 * 3;
    const float cx = xb[p * 3 + 0], cy = xb[p * 3 + 1], cz = xb[p * 3 + 2];
    const float* w = W1 + o * 6;
    const float w0 = w[0], w1 = w[1], w2 = w[2], w3 = w[3], w4 = w[4], w5 = w[5];
    const float gs = g1[o] * BN_INVC, bo = b1[o];
    const int* ib = idx + ((size_t)b * 1024 + p) * 20;
    float best = -3.4e38f;
    for (int j = 0; j < 20; ++j) {
        int nb = ib[j];
        float ex = xb[nb * 3 + 0] - cx, ey = xb[nb * 3 + 1] - cy, ez = xb[nb * 3 + 2] - cz;
        float s = w0 * ex + w1 * ey + w2 * ez + w3 * cx + w4 * cy + w5 * cz;
        best = fmaxf(best, lrelu(s * gs + bo));
    }
    x1[((size_t)b * 1024 + p) * 64 + o] = best;
}

// ---------------- WMMA KNN: pd = 2*XqXc^T - |q|^2 - |c|^2 ----------------------
template <int C>
__global__ __launch_bounds__(256) void knn_wmma_kernel(const float* __restrict__ xin, int* __restrict__ idxout) {
    constexpr int KB = C / 32;
    const int b = blockIdx.y, qbase = blockIdx.x * 16;
    const int tid = threadIdx.x, lane = tid & 31, wave = tid >> 5;
    __shared__ __attribute__((aligned(32))) float dist[16][1024];
    __shared__ __attribute__((aligned(32))) _Float16 a_pk[KB][32][16];
    __shared__ __attribute__((aligned(32))) _Float16 b_pk[8][KB][32][16];
    __shared__ float sqq[16];
    __shared__ float sqc[1024];
    const float* xb = xin + (size_t)b * 1024 * C;

    for (int e = tid; e < 16 * C; e += 256) {           // stage query tile, A-packed
        int m = e / C, k = e % C;
        float f = xb[(qbase + m) * C + k];
        int kb = k >> 5, kl = k & 31;
        a_pk[kb][m + ((kl >> 4) << 4)][kl & 15] = (_Float16)f;
    }
    if (tid < 16) {
        float s = 0.f;
        for (int k = 0; k < C; ++k) { float f = xb[(qbase + tid) * C + k]; s += f * f; }
        sqq[tid] = s;
    }
    for (int c = tid; c < 1024; c += 256) {
        float s = 0.f;
        for (int k = 0; k < C; ++k) { float f = xb[c * C + k]; s += f * f; }
        sqc[c] = s;
    }
    __syncthreads();

    for (int ct = wave; ct < 64; ct += 8) {
        const int cb = ct * 16;
        const int n = lane & 15, khi = (lane >> 4) << 4;
#pragma unroll
        for (int kb = 0; kb < KB; ++kb) {               // stage candidate tile, B-packed
            const float* src = xb + (cb + n) * C + kb * 32 + khi;
#pragma unroll
            for (int s = 0; s < 16; ++s) b_pk[wave][kb][lane][s] = (_Float16)src[s];
        }
        v8f acc = {};
#pragma unroll
        for (int kb = 0; kb < KB; ++kb) {
            v16h a = *(const v16h*)&a_pk[kb][lane][0];
            v16h bb = *(const v16h*)&b_pk[wave][kb][lane][0];
            acc = __builtin_amdgcn_wmma_f32_16x16x32_f16(false, a, false, bb, (short)0, acc, false, false);
        }
        const int nn = lane & 15, mhi = (lane >> 4) << 3;
#pragma unroll
        for (int r = 0; r < 8; ++r) {
            int m = r + mhi;
            dist[m][cb + nn] = 2.f * acc[r] - sqq[m] - sqc[cb + nn];
        }
    }
    __syncthreads();
    for (int rr = wave; rr < 16; rr += 8)
        wave_topk20(&dist[rr][0], idxout + ((size_t)b * 1024 + qbase + rr) * 20);
}

// ---------------- WMMA EdgeConv: gather -> GEMM -> BN/LReLU -> max over k ------
template <int Cin, int Cout>
__global__ __launch_bounds__(256) void edgeconv_wmma_kernel(const float* __restrict__ xin, const int* __restrict__ idx,
                                                            const _Float16* __restrict__ wpk,
                                                            const float* __restrict__ g, const float* __restrict__ bias,
                                                            float* __restrict__ xout) {
    constexpr int K = 2 * Cin;
    constexpr int KB = K / 32;
    constexpr int NT = Cout / 16;
    constexpr int P = 8;
    constexpr int ROWS = P * 20;      // 160
    constexpr int MT = ROWS / 16;     // 10
    const int b = blockIdx.y, pbase = blockIdx.x * P;
    const int tid = threadIdx.x, lane = tid & 31, wave = tid >> 5;
    __shared__ __attribute__((aligned(32))) _Float16 a_pk[MT][KB][32][16];
    __shared__ unsigned omax[P][Cout];
    const float* xb = xin + (size_t)b * 1024 * Cin;
    const int* ib = idx + ((size_t)b * 1024 + pbase) * 20;

    for (int e = tid; e < P * Cout; e += 256) ((unsigned*)omax)[e] = 0u;
    for (int e = tid; e < ROWS * K; e += 256) {         // build edge features, A-packed
        int m = e / K, k = e % K;
        int pl = m / 20, j = m % 20;
        int nb = ib[pl * 20 + j];
        float ctr = xb[(pbase + pl) * Cin + (k < Cin ? k : k - Cin)];
        float val = (k < Cin) ? (xb[nb * Cin + k] - ctr) : ctr;
        int mt = m >> 4, ml = m & 15, kb = k >> 5, kl = k & 31;
        a_pk[mt][kb][ml + ((kl >> 4) << 4)][kl & 15] = (_Float16)val;
    }
    __syncthreads();

    for (int t = wave; t < MT * NT; t += 8) {
        const int mt = t / NT, nt = t % NT;
        v8f acc = {};
        const _Float16* wb = wpk + (size_t)(nt * KB) * 32 * 16;
#pragma unroll
        for (int kb = 0; kb < KB; ++kb) {
            v16h a = *(const v16h*)&a_pk[mt][kb][lane][0];
            v16h wv = *(const v16h*)(wb + ((size_t)kb * 32 + lane) * 16);
            acc = __builtin_amdgcn_wmma_f32_16x16x32_f16(false, a, false, wv, (short)0, acc, false, false);
        }
        const int o = nt * 16 + (lane & 15);
        const float gs = g[o] * BN_INVC, bo = bias[o];
        const int mhi = (lane >> 4) << 3;
#pragma unroll
        for (int r = 0; r < 8; ++r) {
            int m = mt * 16 + r + mhi;
            int pl = m / 20;
            atomicMax(&omax[pl][o], enc_f(lrelu(acc[r] * gs + bo)));
        }
    }
    __syncthreads();
    for (int e = tid; e < P * Cout; e += 256) {
        int pl = e / Cout, o = e % Cout;
        xout[((size_t)b * 1024 + pbase + pl) * Cout + o] = dec_f(omax[pl][o]);
    }
}

// ---------------- conv5: concat(x1..x4) @ W5^T -> BN/LReLU -> global max pool --
__global__ __launch_bounds__(256) void conv5_kernel(const float* __restrict__ x1, const float* __restrict__ x2,
                                                    const float* __restrict__ x3, const float* __restrict__ x4,
                                                    const _Float16* __restrict__ w5pk,
                                                    const float* __restrict__ g5, const float* __restrict__ b5,
                                                    unsigned* __restrict__ pooled) {
    const int b = blockIdx.y, mbase = blockIdx.x * 16;
    const int tid = threadIdx.x, lane = tid & 31, wave = tid >> 5;
    __shared__ __attribute__((aligned(32))) _Float16 a_pk[16][32][16];   // KB = 512/32 = 16
    for (int e = tid; e < 16 * 512; e += 256) {
        int m = e >> 9, k = e & 511;
        int p = mbase + m;
        float f;
        if (k < 64)       f = x1[((size_t)b * 1024 + p) * 64 + k];
        else if (k < 128) f = x2[((size_t)b * 1024 + p) * 64 + (k - 64)];
        else if (k < 256) f = x3[((size_t)b * 1024 + p) * 128 + (k - 128)];
        else              f = x4[((size_t)b * 1024 + p) * 256 + (k - 256)];
        int kb = k >> 5, kl = k & 31;
        a_pk[kb][m + ((kl >> 4) << 4)][kl & 15] = (_Float16)f;
    }
    __syncthreads();
    for (int nt = wave; nt < 32; nt += 8) {
        v8f acc = {};
#pragma unroll
        for (int kb = 0; kb < 16; ++kb) {
            v16h a = *(const v16h*)&a_pk[kb][lane][0];
            v16h wv = *(const v16h*)(w5pk + ((size_t)(nt * 16 + kb) * 32 + lane) * 16);
            acc = __builtin_amdgcn_wmma_f32_16x16x32_f16(false, a, false, wv, (short)0, acc, false, false);
        }
        const int o = nt * 16 + (lane & 15);
        const float gs = g5[o] * BN_INVC, bo = b5[o];
        float best = -3.4e38f;
#pragma unroll
        for (int r = 0; r < 8; ++r) best = fmaxf(best, lrelu(acc[r] * gs + bo));
        atomicMax(&pooled[b * 512 + o], enc_f(best));
    }
}

// ---------------- FC head (tiny, scalar) ---------------------------------------
__global__ __launch_bounds__(256) void fc_head_kernel(const unsigned* __restrict__ pooled,
                                                      const float* fc1w, const float* fc1b, const float* bng1, const float* bnb1,
                                                      const float* fc2w, const float* fc2b, const float* bng2, const float* bnb2,
                                                      const float* fc3w, const float* fc3b, float* __restrict__ out) {
    __shared__ float yv[8][512];
    __shared__ float h1[8][256];
    __shared__ float h2[8][128];
    const int tid = threadIdx.x;
    for (int e = tid; e < 8 * 512; e += 256) ((float*)yv)[e] = dec_f(pooled[e]);
    __syncthreads();
    for (int e = tid; e < 8 * 256; e += 256) {
        int b = e >> 8, o = e & 255;
        float s = fc1b[o];
        const float* w = fc1w + (size_t)o * 512;
        for (int k = 0; k < 512; ++k) s += yv[b][k] * w[k];
        h1[b][o] = lrelu(s * (bng1[o] * BN_INVC) + bnb1[o]);
    }
    __syncthreads();
    for (int e = tid; e < 8 * 128; e += 256) {
        int b = e >> 7, o = e & 127;
        float s = fc2b[o];
        const float* w = fc2w + (size_t)o * 256;
        for (int k = 0; k < 256; ++k) s += h1[b][k] * w[k];
        h2[b][o] = lrelu(s * (bng2[o] * BN_INVC) + bnb2[o]);
    }
    __syncthreads();
    for (int e = tid; e < 8 * 40; e += 256) {
        int b = e / 40, o = e % 40;
        float s = fc3b[o];
        const float* w = fc3w + (size_t)o * 128;
        for (int k = 0; k < 128; ++k) s += h2[b][k] * w[k];
        out[b * 40 + o] = s;
    }
}

// ---------------- launch --------------------------------------------------------
extern "C" void kernel_launch(void* const* d_in, const int* in_sizes, int n_in,
                              void* d_out, int out_size, void* d_ws, size_t ws_size,
                              hipStream_t stream) {
    (void)in_sizes; (void)n_in; (void)out_size; (void)ws_size;
    const float* x    = (const float*)d_in[0];
    const float* W1   = (const float*)d_in[1];
    const float* g1   = (const float*)d_in[2];
    const float* b1   = (const float*)d_in[3];
    const float* W2   = (const float*)d_in[4];
    const float* g2   = (const float*)d_in[5];
    const float* b2   = (const float*)d_in[6];
    const float* W3   = (const float*)d_in[7];
    const float* g3   = (const float*)d_in[8];
    const float* b3   = (const float*)d_in[9];
    const float* W4   = (const float*)d_in[10];
    const float* g4   = (const float*)d_in[11];
    const float* b4   = (const float*)d_in[12];
    const float* W5   = (const float*)d_in[13];
    const float* g5   = (const float*)d_in[14];
    const float* b5   = (const float*)d_in[15];
    const float* fc1w = (const float*)d_in[16];
    const float* fc1b = (const float*)d_in[17];
    const float* bng1 = (const float*)d_in[18];
    const float* bnb1 = (const float*)d_in[19];
    const float* fc2w = (const float*)d_in[20];
    const float* fc2b = (const float*)d_in[21];
    const float* bng2 = (const float*)d_in[22];
    const float* bnb2 = (const float*)d_in[23];
    const float* fc3w = (const float*)d_in[24];
    const float* fc3b = (const float*)d_in[25];

    char* ws = (char*)d_ws;
    size_t off = 0;
    auto take = [&](size_t bytes) -> char* {
        char* p = ws + off;
        off += (bytes + 255) & ~(size_t)255;
        return p;
    };
    float*    x1     = (float*)take(8ull * 1024 * 64 * 4);
    float*    x2     = (float*)take(8ull * 1024 * 64 * 4);
    float*    x3     = (float*)take(8ull * 1024 * 128 * 4);
    float*    x4     = (float*)take(8ull * 1024 * 256 * 4);
    int*      idx    = (int*)take(8ull * 1024 * 20 * 4);
    unsigned* pooled = (unsigned*)take(8ull * 512 * 4);
    _Float16* p2     = (_Float16*)take(64ull * 128 * 2);
    _Float16* p3     = (_Float16*)take(128ull * 128 * 2);
    _Float16* p4     = (_Float16*)take(256ull * 256 * 2);
    _Float16* p5     = (_Float16*)take(512ull * 512 * 2);

    pack_weights_kernel<<<dim3(64), 256, 0, stream>>>(W2, W3, W4, W5, p2, p3, p4, p5);
    init_pooled_kernel<<<dim3(16), 256, 0, stream>>>(pooled);

    knn3_kernel<<<dim3(64, 8), 256, 0, stream>>>(x, idx);
    edgeconv1_kernel<<<dim3(256, 8), 256, 0, stream>>>(x, idx, W1, g1, b1, x1);

    knn_wmma_kernel<64><<<dim3(64, 8), 256, 0, stream>>>(x1, idx);
    edgeconv_wmma_kernel<64, 64><<<dim3(128, 8), 256, 0, stream>>>(x1, idx, p2, g2, b2, x2);

    knn_wmma_kernel<64><<<dim3(64, 8), 256, 0, stream>>>(x2, idx);
    edgeconv_wmma_kernel<64, 128><<<dim3(128, 8), 256, 0, stream>>>(x2, idx, p3, g3, b3, x3);

    knn_wmma_kernel<128><<<dim3(64, 8), 256, 0, stream>>>(x3, idx);
    edgeconv_wmma_kernel<128, 256><<<dim3(128, 8), 256, 0, stream>>>(x3, idx, p4, g4, b4, x4);

    conv5_kernel<<<dim3(64, 8), 256, 0, stream>>>(x1, x2, x3, x4, p5, g5, b5, pooled);
    fc_head_kernel<<<dim3(1), 256, 0, stream>>>(pooled, fc1w, fc1b, bng1, bnb1,
                                                fc2w, fc2b, bng2, bnb2, fc3w, fc3b, (float*)d_out);
}
